// TemporalContrastHead_25366076850430
// MI455X (gfx1250) — matrix operand order repaired and checked by reading, not compiled, not run on hardware
//
#include <hip/hip_runtime.h>
#include <hip/hip_bf16.h>
#include <math.h>

// ---------------------------------------------------------------------------
// TemporalContrastHead for MI455X (gfx1250): bf16 WMMA projections + fused
// similarity/loss sweep that never materializes the (B,T,T) sims tensor.
// B=32, T=1024, D=256. wave32, V_WMMA_F32_16X16X32_BF16.
// Sims kernel: M=32 rows per wave -> 16 WMMA per 8 B-fragment loads (2x the
// matrix-op-per-byte of a single 16x16 tile), halving Pv L2 re-read traffic.
// ---------------------------------------------------------------------------

typedef __attribute__((ext_vector_type(16))) __bf16 v16bf;
typedef __attribute__((ext_vector_type(8)))  __bf16 v8bf;
typedef __attribute__((ext_vector_type(8)))  float  v8f;

#define TC_B 32
#define TC_T 1024
#define TC_D 256
#define TC_INV_TEMP (1.0f / 0.07f)

static __device__ __forceinline__ v16bf tc_cat8(v8bf lo, v8bf hi) {
  return __builtin_shufflevector(lo, hi, 0,1,2,3,4,5,6,7,8,9,10,11,12,13,14,15);
}

// ---------------------------------------------------------------------------
// Kernel 1: per-batch valid mask, compressed rank (cumsum-1), Tv, zero accum.
// ---------------------------------------------------------------------------
__global__ __launch_bounds__(32)
void tc_rank_init_kernel(const int* __restrict__ mask, int* __restrict__ rank,
                         int* __restrict__ validA, int* __restrict__ Tv,
                         float* __restrict__ bloss) {
  const int b = threadIdx.x;
  if (b < TC_B) {
    const int* mrow = mask + b * TC_T;
    int* rrow = rank + b * TC_T;
    int* vrow = validA + b * TC_T;
    int cnt = 0;
    for (int t = 0; t < TC_T; ++t) {
      int v = mrow[t] > 0;
      cnt += v;
      rrow[t] = cnt - 1;
      vrow[t] = v;
    }
    Tv[b] = cnt;
    bloss[b] = 0.0f;
  }
}

// ---------------------------------------------------------------------------
// Kernel 2: convert both weight matrices (row-major [e][d]) to bf16.
// ---------------------------------------------------------------------------
__global__ void tc_wconv_kernel(const float* __restrict__ W0,
                                const float* __restrict__ W1,
                                __bf16* __restrict__ B0,
                                __bf16* __restrict__ B1) {
  int i = blockIdx.x * blockDim.x + threadIdx.x;
  if (i < TC_D * TC_D) {
    B0[i] = (__bf16)W0[i];
    B1[i] = (__bf16)W1[i];
  }
}

// ---------------------------------------------------------------------------
// Kernel 3: projection y = x @ W^T (WMMA bf16, f32 acc) + L2 normalize,
// emitting bf16 embeddings. One wave per 16-row tile of flattened (B*T) rows.
// ---------------------------------------------------------------------------
__global__ __launch_bounds__(32)
void tc_proj_norm_kernel(const float* __restrict__ X,
                         const __bf16* __restrict__ Wb,
                         __bf16* __restrict__ P) {
  __shared__ __align__(32) __bf16 xb[16 * TC_D];  // 8 KB: bf16 input tile
  __shared__ __align__(32) float  yl[16 * TC_D];  // 16 KB: f32 projected tile

  const int lane = threadIdx.x;
  const long row0 = (long)blockIdx.x * 16;

  // stage input tile into LDS as bf16
  for (int i = lane; i < 16 * TC_D; i += 32) {
    int r = i >> 8, c = i & 255;
    xb[i] = (__bf16)X[(row0 + r) * TC_D + c];
  }
  __syncthreads();

  const int m = lane & 15;
  const int half = lane >> 4;

  // A fragments for all 8 K-steps (contiguous K, matching B):
  // lanes 0-15: row M=m, K = ks*32+0..15
  // lanes 16-31: row M=m, K = ks*32+16..31
  v16bf A[8];
#pragma unroll
  for (int ks = 0; ks < 8; ++ks) {
    const __bf16* ap = &xb[m * TC_D + ks * 32 + half * 16];
    v8bf lo = *(const v8bf*)(ap);
    v8bf hi = *(const v8bf*)(ap + 8);
    A[ks] = tc_cat8(lo, hi);
  }

  // sweep 16 output-column tiles
  for (int j0 = 0; j0 < TC_D; j0 += 16) {
    v8f c = {};
#pragma unroll
    for (int ks = 0; ks < 8; ++ks) {
      // B fragment: lane holds column N = j0+m of W^T == row (j0+m) of W,
      // K = ks*32 + (half ? 16..31 : 0..15): one contiguous 32B load.
      const __bf16* bp = &Wb[(long)(j0 + m) * TC_D + ks * 32 + half * 16];
      v16bf bfrag = *(const v16bf*)bp;
      c = __builtin_amdgcn_wmma_f32_16x16x32_bf16(false, A[ks], false, bfrag,
                                                  (short)0, c, false, false);
    }
    // C layout: VGPR r, lanes 0-15 -> (M=r, N=m); lanes 16-31 -> (M=8+r, N=m)
#pragma unroll
    for (int r = 0; r < 8; ++r)
      yl[(half * 8 + r) * TC_D + j0 + m] = c[r];
  }
  __syncthreads();

  // L2 normalize: lanes (m, m+16) co-own row m, each handles 128 columns
  const float* yr = &yl[m * TC_D + half * 128];
  float ssq = 0.0f;
#pragma unroll 8
  for (int cdx = 0; cdx < 128; ++cdx) {
    float t = yr[cdx];
    ssq += t * t;
  }
  ssq += __shfl_xor(ssq, 16, 32);
  float scale = 1.0f / fmaxf(sqrtf(ssq), 1e-12f);
  __bf16* pr = &P[(row0 + m) * TC_D + half * 128];
#pragma unroll 8
  for (int cdx = 0; cdx < 128; ++cdx)
    pr[cdx] = (__bf16)(yr[cdx] * scale);
}

// ---------------------------------------------------------------------------
// Kernel 4: fused sims sweep + banded-positive max + logsumexp + loss accum.
// One wave per (batch, 32-row i-tile): two 16x16 C tiles share each B
// fragment (16 WMMA : 8 loads). Sweeps 64 j-tiles x 8 WMMA K-steps each.
// ---------------------------------------------------------------------------
__global__ __launch_bounds__(32)
void tc_sims_loss_kernel(const __bf16* __restrict__ Pe,
                         const __bf16* __restrict__ Pv,
                         const int* __restrict__ rank,
                         const int* __restrict__ validA,
                         float* __restrict__ bloss) {
  __shared__ int rs[TC_T];
  __shared__ int vs[TC_T];

  const int lane = threadIdx.x;
  const int b = blockIdx.x >> 5;          // 32 i-tiles (of 32 rows) per batch
  const int i0 = (blockIdx.x & 31) << 5;  // row tile base within batch
  const long base = (long)b * TC_T;

  for (int t = lane; t < TC_T; t += 32) {
    rs[t] = rank[base + t];
    vs[t] = validA[base + t];
  }
  __syncthreads();

  const int m = lane & 15;
  const int half = lane >> 4;

  // per-lane row metadata: C-tile u (u=0,1) rows are i0 + u*16 + half*8 + r
  int rI[2][8], vI[2][8];
#pragma unroll
  for (int u = 0; u < 2; ++u)
#pragma unroll
    for (int r = 0; r < 8; ++r) {
      int row = i0 + u * 16 + half * 8 + r;
      rI[u][r] = rs[row];
      vI[u][r] = vs[row];
    }

  // preload A fragments for both 16-row tiles (32 rows of Pe, full K=256)
  v16bf A0[8], A1[8];
  {
    const __bf16* ar0 = Pe + (base + i0 + m) * TC_D;
    const __bf16* ar1 = Pe + (base + i0 + 16 + m) * TC_D;
#pragma unroll
    for (int ks = 0; ks < 8; ++ks) {
      const __bf16* ap0 = ar0 + ks * 32 + half * 16;
      const __bf16* ap1 = ar1 + ks * 32 + half * 16;
      A0[ks] = tc_cat8(*(const v8bf*)(ap0), *(const v8bf*)(ap0 + 8));
      A1[ks] = tc_cat8(*(const v8bf*)(ap1), *(const v8bf*)(ap1 + 8));
    }
  }

  float pm[2][8], es[2][8];
#pragma unroll
  for (int u = 0; u < 2; ++u)
#pragma unroll
    for (int r = 0; r < 8; ++r) { pm[u][r] = -1.0e30f; es[u][r] = 0.0f; }

#pragma unroll 1
  for (int j0 = 0; j0 < TC_T; j0 += 16) {
    v8f c0 = {}, c1 = {};
    const __bf16* brow = Pv + (base + j0 + m) * TC_D;
#pragma unroll
    for (int ks = 0; ks < 8; ++ks) {
      // B fragment: lane holds column j0+m of sims (= row j0+m of Pv),
      // K = ks*32 + (half ? 16..31 : 0..15); reused by both C tiles.
      v16bf bfrag = *(const v16bf*)(brow + ks * 32 + half * 16);
      c0 = __builtin_amdgcn_wmma_f32_16x16x32_bf16(false, A0[ks], false, bfrag,
                                                   (short)0, c0, false, false);
      c1 = __builtin_amdgcn_wmma_f32_16x16x32_bf16(false, A1[ks], false, bfrag,
                                                   (short)0, c1, false, false);
    }
    // epilogue: fold tiles into running per-row (max over positives, sum exp)
    const int col = j0 + m;
    const int vJ = vs[col];
    const int rJ = rs[col];
#pragma unroll
    for (int r = 0; r < 8; ++r) {
      float s0 = c0[r] * TC_INV_TEMP;         // |s| <= 14.3 -> exp safe in f32
      bool p0 = (vI[0][r] != 0) && (vJ != 0);
      es[0][r] += p0 ? __expf(s0) : 0.0f;
      int d0 = rI[0][r] - rJ; d0 = d0 < 0 ? -d0 : d0;
      if (p0 && d0 <= 2) pm[0][r] = fmaxf(pm[0][r], s0);

      float s1 = c1[r] * TC_INV_TEMP;
      bool p1 = (vI[1][r] != 0) && (vJ != 0);
      es[1][r] += p1 ? __expf(s1) : 0.0f;
      int d1 = rI[1][r] - rJ; d1 = d1 < 0 ? -d1 : d1;
      if (p1 && d1 <= 2) pm[1][r] = fmaxf(pm[1][r], s1);
    }
  }

  // reduce across the 16-lane group owning each row half
#pragma unroll
  for (int u = 0; u < 2; ++u)
#pragma unroll
    for (int r = 0; r < 8; ++r) {
#pragma unroll
      for (int off = 1; off < 16; off <<= 1) {
        es[u][r] += __shfl_xor(es[u][r], off, 32);
        pm[u][r] = fmaxf(pm[u][r], __shfl_xor(pm[u][r], off, 32));
      }
    }

  if (m == 0) {  // lane 0 / lane 16 hold finished rows for their half
    float lsum = 0.0f;
#pragma unroll
    for (int u = 0; u < 2; ++u)
#pragma unroll
      for (int r = 0; r < 8; ++r)
        if (vI[u][r]) lsum += __logf(es[u][r]) - pm[u][r];  // lse - pos
    atomicAdd(&bloss[b], lsum);
  }
}

// ---------------------------------------------------------------------------
// Kernel 5: final masked mean over batches.
// ---------------------------------------------------------------------------
__global__ __launch_bounds__(32)
void tc_final_kernel(const float* __restrict__ bloss, const int* __restrict__ Tv,
                     float* __restrict__ out) {
  const int b = threadIdx.x;
  float pb = 0.0f;
  if (b < TC_B) {
    int tv = Tv[b];
    if (tv >= 2) pb = bloss[b] / (float)tv;
  }
#pragma unroll
  for (int off = 16; off > 0; off >>= 1) pb += __shfl_xor(pb, off, 32);
  if (b == 0) out[0] = pb / ((float)TC_B + 1e-8f);
}

// ---------------------------------------------------------------------------
// Host launch
// ---------------------------------------------------------------------------
extern "C" void kernel_launch(void* const* d_in, const int* in_sizes, int n_in,
                              void* d_out, int out_size, void* d_ws, size_t ws_size,
                              hipStream_t stream) {
  const float* eeg  = (const float*)d_in[0];   // (32,1024,256) f32
  const float* eye  = (const float*)d_in[1];   // (32,1024,256) f32
  const int*   mask = (const int*)  d_in[2];   // (32,1024) i32
  const float* Weeg = (const float*)d_in[3];   // (256,256) f32
  const float* Weye = (const float*)d_in[4];   // (256,256) f32
  float* out = (float*)d_out;

  char* ws = (char*)d_ws;
  const size_t embBytes = (size_t)TC_B * TC_T * TC_D * sizeof(__bf16);  // 16 MB
  const size_t idxBytes = (size_t)TC_B * TC_T * sizeof(int);            // 128 KB
  const size_t wBytes   = (size_t)TC_D * TC_D * sizeof(__bf16);         // 128 KB
  __bf16* Pe     = (__bf16*)(ws);
  __bf16* Pv     = (__bf16*)(ws + embBytes);
  int*    rank   = (int*)   (ws + 2 * embBytes);
  int*    validA = (int*)   (ws + 2 * embBytes + idxBytes);
  __bf16* Wbe    = (__bf16*)(ws + 2 * embBytes + 2 * idxBytes);
  __bf16* Wbv    = (__bf16*)(ws + 2 * embBytes + 2 * idxBytes + wBytes);
  int*    Tv     = (int*)   (ws + 2 * embBytes + 2 * idxBytes + 2 * wBytes);
  float*  bloss  = (float*) (ws + 2 * embBytes + 2 * idxBytes + 2 * wBytes + 128);

  const int rowTiles = (TC_B * TC_T) / 16;  // 2048

  tc_rank_init_kernel<<<1, 32, 0, stream>>>(mask, rank, validA, Tv, bloss);
  tc_wconv_kernel<<<(TC_D * TC_D + 255) / 256, 256, 0, stream>>>(Weeg, Weye, Wbe, Wbv);
  tc_proj_norm_kernel<<<rowTiles, 32, 0, stream>>>(eeg, Wbe, Pe);
  tc_proj_norm_kernel<<<rowTiles, 32, 0, stream>>>(eye, Wbv, Pv);
  tc_sims_loss_kernel<<<TC_B * (TC_T / 32), 32, 0, stream>>>(Pe, Pv, rank, validA, bloss);
  tc_final_kernel<<<1, 32, 0, stream>>>(bloss, Tv, out);
}
